// GraphNet_24661702213865
// MI455X (gfx1250) — matrix operand order repaired and checked by reading, not compiled, not run on hardware
//
#include <hip/hip_runtime.h>
#include <math.h>

// ---------------------------------------------------------------------------
// 3-layer GCN + global_add_pool for MI455X (gfx1250, wave32).
//  - GEMM h = x@W via v_wmma_f32_16x16x32_bf16 (bf16 in, f32 accum)
//  - edge scatter-add via global_atomic_add_f32 (L2 resident: 25.6MB << 192MB)
// NNODES is a multiple of 16, so the 16-row GEMM tiles need no bounds checks.
// ---------------------------------------------------------------------------

#define NNODES  50000
#define NGRAPHS 512
#define DFEAT   128

typedef __attribute__((ext_vector_type(16))) __bf16 v16bf;
typedef __attribute__((ext_vector_type(8)))  __bf16 v8bf;
typedef __attribute__((ext_vector_type(8)))  float  v8f;

// ---------------- utility kernels ----------------
__global__ void zero_f(float* p, int n) {
    int i = blockIdx.x * blockDim.x + threadIdx.x;
    if (i < n) p[i] = 0.0f;
}

__global__ void count_deg(const int* __restrict__ dst, float* __restrict__ deg, int nE) {
    int i = blockIdx.x * blockDim.x + threadIdx.x;
    if (i < nE) atomicAdd(&deg[dst[i]], 1.0f);
}

// dis[i] = rsqrt(indeg + 1)   (self-loop guarantees deg >= 1, so no branch)
__global__ void finalize_dis(float* d, int n) {
    int i = blockIdx.x * blockDim.x + threadIdx.x;
    if (i < n) d[i] = rsqrtf(d[i] + 1.0f);
}

// ---------------- WMMA GEMM: H[n,128] = X[n,128] @ W[128,128] ----------------
// Block = 256 threads = 8 waves. Block handles 16 rows; each wave owns a
// 16-column tile. K=128 swept as 4 x wmma_f32_16x16x32_bf16.
#define LDSP 136   // padded row stride (elements) to avoid LDS bank conflicts

__global__ void gemm_bf16_wmma(const float* __restrict__ X,
                               const float* __restrict__ W,
                               float* __restrict__ H) {
    __shared__ __align__(16) __bf16 sWt[DFEAT * LDSP];  // W transposed: [col][k]
    __shared__ __align__(16) __bf16 sX [16    * LDSP];  // 16-row tile of X

    const int tid     = threadIdx.x;
    const int rowBase = blockIdx.x * 16;

    // Stage W (transposed) as bf16 via float4 loads:
    //   sWt[c*LDSP + k] = W[k*128 + c]
    for (int i = tid; i < (DFEAT * DFEAT) / 4; i += 256) {
        const float4 w4 = ((const float4*)W)[i];
        const int k = (i * 4) >> 7, c = (i * 4) & 127;
        sWt[(c + 0) * LDSP + k] = (__bf16)w4.x;
        sWt[(c + 1) * LDSP + k] = (__bf16)w4.y;
        sWt[(c + 2) * LDSP + k] = (__bf16)w4.z;
        sWt[(c + 3) * LDSP + k] = (__bf16)w4.w;
    }
    // Stage 16-row X tile as bf16 via float4 loads (rows always in range).
    const float4* X4 = (const float4*)(X + (size_t)rowBase * DFEAT);
    for (int i = tid; i < (16 * DFEAT) / 4; i += 256) {
        const float4 v = X4[i];
        const int r = (i * 4) >> 7, c = (i * 4) & 127;
        sX[r * LDSP + c + 0] = (__bf16)v.x;
        sX[r * LDSP + c + 1] = (__bf16)v.y;
        sX[r * LDSP + c + 2] = (__bf16)v.z;
        sX[r * LDSP + c + 3] = (__bf16)v.w;
    }
    __syncthreads();

    const int wave    = tid >> 5;          // 0..7 -> column tile
    const int lane    = tid & 31;
    const int colBase = wave * 16;
    const int mn      = lane & 15;         // row (A) / col (B/C/D) within tile
    const int half    = lane >> 4;

    v8f acc = {};
#pragma unroll
    for (int k0 = 0; k0 < DFEAT; k0 += 32) {
        // A fragment (16x32 bf16): elem j -> K = k0 + j + 8*half (j<8),
        //                                   K = k0 + j + 8 + 8*half (j>=8)
        v8bf a0 = *(const v8bf*)(sX + mn * LDSP + k0 + 8 * half);
        v8bf a1 = *(const v8bf*)(sX + mn * LDSP + k0 + 16 + 8 * half);
        // B fragment (32x16 bf16): same K mapping as A, N = colBase+mn
        v8bf b0 = *(const v8bf*)(sWt + (colBase + mn) * LDSP + k0 + 8 * half);
        v8bf b1 = *(const v8bf*)(sWt + (colBase + mn) * LDSP + k0 + 16 + 8 * half);

        v16bf a, b;
#pragma unroll
        for (int j = 0; j < 8; ++j) {
            a[j] = a0[j]; a[j + 8] = a1[j];
            b[j] = b0[j]; b[j + 8] = b1[j];
        }
        acc = __builtin_amdgcn_wmma_f32_16x16x32_bf16(
            /*neg_a=*/false, a, /*neg_b=*/false, b,
            /*c_mod=*/(short)0, acc, /*reuse_a=*/false, /*reuse_b=*/false);
    }

    // D layout: VGPR r, lane -> M = r + 8*half, N = mn. No bounds check needed.
    float* o = H + (size_t)(rowBase + 8 * half) * DFEAT + colBase + mn;
#pragma unroll
    for (int r = 0; r < 8; ++r)
        o[r * DFEAT] = acc[r];
}

// out[i][:] = bias + dis[i]^2 * h[i][:]    (bias + self-loop term)
__global__ void init_out_k(const float* __restrict__ H, const float* __restrict__ dis,
                           const float* __restrict__ bias, float* __restrict__ out,
                           int nelem) {
    int i = blockIdx.x * blockDim.x + threadIdx.x;
    if (i >= nelem) return;
    int node = i >> 7, c = i & 127;
    float s = dis[node];
    out[i] = bias[c] + s * s * H[i];
}

// One wave per edge: lanes cover 128 floats as float4 each. Gather coalesced,
// scatter via f32 atomics (device scope, lands in L2).
__global__ void scatter_edges(const float* __restrict__ H, const float* __restrict__ dis,
                              const int* __restrict__ src, const int* __restrict__ dst,
                              float* __restrict__ out, int nE) {
    int gid = blockIdx.x * blockDim.x + threadIdx.x;
    if (gid >= nE * 32) return;
    int e = gid >> 5;
    int c = (gid & 31) * 4;
    int s = src[e], d = dst[e];
    float nrm = dis[s] * dis[d];
    const float4 hv = *(const float4*)(H + (size_t)s * DFEAT + c);
    float* o = out + (size_t)d * DFEAT + c;
    atomicAdd(o + 0, nrm * hv.x);
    atomicAdd(o + 1, nrm * hv.y);
    atomicAdd(o + 2, nrm * hv.z);
    atomicAdd(o + 3, nrm * hv.w);
}

__global__ void relu_k(float* p, int n) {
    int i = blockIdx.x * blockDim.x + threadIdx.x;
    if (i < n) p[i] = fmaxf(p[i], 0.0f);
}

// global_add_pool: one wave per node, float4 atomics into [512,128]
__global__ void pool_k(const float* __restrict__ x, const int* __restrict__ batch,
                       float* __restrict__ out, int n) {
    int gid = blockIdx.x * blockDim.x + threadIdx.x;
    if (gid >= n * 32) return;
    int node = gid >> 5;
    int c = (gid & 31) * 4;
    int g = batch[node];
    const float4 v = *(const float4*)(x + (size_t)node * DFEAT + c);
    float* o = out + (size_t)g * DFEAT + c;
    atomicAdd(o + 0, v.x);
    atomicAdd(o + 1, v.y);
    atomicAdd(o + 2, v.z);
    atomicAdd(o + 3, v.w);
}

extern "C" void kernel_launch(void* const* d_in, const int* in_sizes, int n_in,
                              void* d_out, int out_size, void* d_ws, size_t ws_size,
                              hipStream_t stream) {
    const float* x  = (const float*)d_in[0];
    const float* Wm[3] = { (const float*)d_in[1], (const float*)d_in[3], (const float*)d_in[5] };
    const float* bv[3] = { (const float*)d_in[2], (const float*)d_in[4], (const float*)d_in[6] };
    const int* ei    = (const int*)d_in[7];
    const int* batch = (const int*)d_in[8];

    const int nE  = in_sizes[7] / 2;        // 800000
    const int* src = ei;
    const int* dst = ei + nE;

    // Workspace layout (floats): dis[50000] | bufA[50000*128] | bufB[50000*128]
    float* ws   = (float*)d_ws;
    float* dis  = ws;
    float* bufA = ws + 51200;               // h (GEMM output)
    float* bufB = bufA + (size_t)NNODES * DFEAT;  // aggregated output / next x

    const int NF = NNODES * DFEAT;          // 6,400,000

    // normalization
    zero_f<<<(NNODES + 255) / 256, 256, 0, stream>>>(dis, NNODES);
    count_deg<<<(nE + 255) / 256, 256, 0, stream>>>(dst, dis, nE);
    finalize_dis<<<(NNODES + 255) / 256, 256, 0, stream>>>(dis, NNODES);

    // 3 GCN layers (ping-pong: h in bufA, aggregated x' in bufB)
    const float* xin = x;
    for (int l = 0; l < 3; ++l) {
        gemm_bf16_wmma<<<NNODES / 16, 256, 0, stream>>>(xin, Wm[l], bufA);
        init_out_k<<<(NF + 255) / 256, 256, 0, stream>>>(bufA, dis, bv[l], bufB, NF);
        scatter_edges<<<(nE * 32 + 255) / 256, 256, 0, stream>>>(bufA, dis, src, dst, bufB, nE);
        relu_k<<<(NF + 255) / 256, 256, 0, stream>>>(bufB, NF);
        xin = bufB;
    }

    // global_add_pool -> d_out [512, 128]
    zero_f<<<(NGRAPHS * DFEAT + 255) / 256, 256, 0, stream>>>((float*)d_out, NGRAPHS * DFEAT);
    pool_k<<<(NNODES * 32 + 255) / 256, 256, 0, stream>>>(bufB, batch, (float*)d_out, NNODES);
}
